// EdgeConv_Block_20899310862676
// MI455X (gfx1250) — matrix operand order repaired and checked by reading, not compiled, not run on hardware
//
#include <hip/hip_runtime.h>
#include <hip/hip_bf16.h>

#define Bx 16
#define Nx 1024
#define Dx 64
#define Cx 64
#define Kx 20
#define EPSx 0.001f

typedef __attribute__((ext_vector_type(16))) __bf16 v16bf;
typedef __attribute__((ext_vector_type(8)))  __bf16 v8bf;
typedef __attribute__((ext_vector_type(8)))  float  v8f;

__device__ __forceinline__ v8f wmma_bf16(v16bf a, v16bf b, v8f c) {
  // D = A(16x32 bf16) * B(32x16 bf16) + C(16x16 f32)
  return __builtin_amdgcn_wmma_f32_16x16x32_bf16(false, a, false, b, (short)0, c, false, false);
}

__device__ __forceinline__ v16bf pack16(float4 a, float4 b, float4 c, float4 d) {
  v16bf r;
  r[0]=(__bf16)a.x;  r[1]=(__bf16)a.y;  r[2]=(__bf16)a.z;  r[3]=(__bf16)a.w;
  r[4]=(__bf16)b.x;  r[5]=(__bf16)b.y;  r[6]=(__bf16)b.z;  r[7]=(__bf16)b.w;
  r[8]=(__bf16)c.x;  r[9]=(__bf16)c.y;  r[10]=(__bf16)c.z; r[11]=(__bf16)c.w;
  r[12]=(__bf16)d.x; r[13]=(__bf16)d.y; r[14]=(__bf16)d.z; r[15]=(__bf16)d.w;
  return r;
}

__device__ __forceinline__ v16bf packdiff16(float4 a, float4 b, float4 c, float4 d,
                                            float4 e, float4 f, float4 g, float4 h) {
  v16bf r;
  r[0]=(__bf16)(a.x-e.x);  r[1]=(__bf16)(a.y-e.y);  r[2]=(__bf16)(a.z-e.z);  r[3]=(__bf16)(a.w-e.w);
  r[4]=(__bf16)(b.x-f.x);  r[5]=(__bf16)(b.y-f.y);  r[6]=(__bf16)(b.z-f.z);  r[7]=(__bf16)(b.w-f.w);
  r[8]=(__bf16)(c.x-g.x);  r[9]=(__bf16)(c.y-g.y);  r[10]=(__bf16)(c.z-g.z); r[11]=(__bf16)(c.w-g.w);
  r[12]=(__bf16)(d.x-h.x); r[13]=(__bf16)(d.y-h.y); r[14]=(__bf16)(d.z-h.z); r[15]=(__bf16)(d.w-h.w);
  return r;
}

__device__ __forceinline__ v16bf cat8(v8bf a, v8bf b) {
  v16bf r;
#pragma unroll
  for (int i = 0; i < 8; ++i) { r[i] = a[i]; r[8 + i] = b[i]; }
  return r;
}

// ---------------------------------------------------------------------------
// prep: f0 gather, BN fold to (scale, shift), repack w1/w2 into bf16 WMMA
// B-fragment order: flat = ((kt*4+nt)*32 + lane)*16 + e, k = kt*32 + (lane/16)*8 + (e%8) + (e/8)*16,
// c = nt*16 + (lane%16).  Lane fragment = 32 contiguous bytes.
// ---------------------------------------------------------------------------
__global__ void prep_kernel(const float* __restrict__ feats,
                            const float* __restrict__ w1, const float* __restrict__ b1,
                            const float* __restrict__ g1, const float* __restrict__ be1,
                            const float* __restrict__ mu1, const float* __restrict__ v1,
                            const float* __restrict__ w2, const float* __restrict__ b2,
                            const float* __restrict__ g2, const float* __restrict__ be2,
                            const float* __restrict__ mu2, const float* __restrict__ v2,
                            float* __restrict__ f0, __bf16* __restrict__ w1p,
                            __bf16* __restrict__ w2p, float* __restrict__ st) {
  int tid = blockIdx.x * 256 + threadIdx.x;
  if (tid < Bx * Nx) f0[tid] = feats[(size_t)tid * Dx];
  if (tid < 8192) {  // w1 pack: 4 kt x 4 nt x 32 lanes x 16 elems
    int e = tid & 15, lane = (tid >> 4) & 31, nt = (tid >> 9) & 3, kt = tid >> 11;
    int k = kt * 32 + (lane >> 4) * 8 + (e & 7) + (e >> 3) * 16;
    int c = nt * 16 + (lane & 15);
    w1p[tid] = (__bf16)w1[k * Cx + c];
  }
  if (tid < 4096) {  // w2 pack: 2 kt x 4 nt x 32 lanes x 16 elems
    int e = tid & 15, lane = (tid >> 4) & 31, nt = (tid >> 9) & 3, kt = tid >> 11;
    int k = kt * 32 + (lane >> 4) * 8 + (e & 7) + (e >> 3) * 16;
    int c = nt * 16 + (lane & 15);
    w2p[tid] = (__bf16)w2[k * Cx + c];
  }
  if (tid < Cx) {
    float s = g1[tid] * rsqrtf(v1[tid] + EPSx);
    st[tid] = s;
    st[64 + tid] = (b1[tid] - mu1[tid]) * s + be1[tid];
    float s2 = g2[tid] * rsqrtf(v2[tid] + EPSx);
    st[128 + tid] = s2;
    st[192 + tid] = (b2[tid] - mu2[tid]) * s2 + be2[tid];
  }
}

// ---------------------------------------------------------------------------
// topk: one wave32 per (b,n) row; lane-local sorted top-20 in registers,
// then 20 rounds of wave argmin merge via shuffles.
// ---------------------------------------------------------------------------
__global__ void topk_kernel(const float* __restrict__ adj,
                            const float* __restrict__ f0,
                            int* __restrict__ nn) {
  const int wave = threadIdx.x >> 5;
  const int lane = threadIdx.x & 31;
  const int row = blockIdx.x * 8 + wave;  // 0..B*N-1
  const int b = row >> 10;
  const int n = row & (Nx - 1);
  const float f0n = f0[b * Nx + n];
  const float* __restrict__ arow = adj + (size_t)row * Nx;
  const float* __restrict__ f0b = f0 + b * Nx;

  float best[Kx];
  int bidx[Kx];
#pragma unroll
  for (int i = 0; i < Kx; ++i) { best[i] = 1e30f; bidx[i] = 0; }

  for (int t = 0; t < Nx / 32; ++t) {
    int j = lane + t * 32;
    float v = arow[j] * fabsf(f0b[j] - f0n);
    if (v < best[Kx - 1]) {
      best[Kx - 1] = v;
      bidx[Kx - 1] = j;
#pragma unroll
      for (int i = Kx - 1; i > 0; --i) {
        bool sw = best[i] < best[i - 1];
        float tv = best[i - 1]; int ti = bidx[i - 1];
        float nv = sw ? best[i] : best[i - 1]; int ni = sw ? bidx[i] : bidx[i - 1];
        best[i]     = sw ? tv : best[i];
        bidx[i]     = sw ? ti : bidx[i];
        best[i - 1] = nv;
        bidx[i - 1] = ni;
      }
    }
  }

  int* __restrict__ out = nn + (size_t)row * Kx;
  for (int r = 0; r < Kx; ++r) {
    float v = best[0]; int ii = bidx[0]; int l = lane;
#pragma unroll
    for (int off = 16; off > 0; off >>= 1) {
      float ov = __shfl_xor(v, off, 32);
      int oi = __shfl_xor(ii, off, 32);
      int ol = __shfl_xor(l, off, 32);
      bool take = (ov < v) || (ov == v && ol < l);
      v = take ? ov : v; ii = take ? oi : ii; l = take ? ol : l;
    }
    if (lane == l) {
      out[r] = ii;
#pragma unroll
      for (int i = 0; i < Kx - 1; ++i) { best[i] = best[i + 1]; bidx[i] = bidx[i + 1]; }
      best[Kx - 1] = 1e30f;
    }
  }
}

// ---------------------------------------------------------------------------
// edgeconv: one wave32 per point. GEMM1 (20x128 * 128x64) + BN/ReLU -> LDS,
// GEMM2 (20x64 * 64x64) + BN/ReLU -> mean over k. nt-outer loop: each set of
// B fragments is loaded once (one clause of b128 loads) and feeds 8 (GEMM1)
// or 4 (GEMM2) back-to-back v_wmma_f32_16x16x32_bf16.
// ---------------------------------------------------------------------------
#define XSTRIDE 72  // bf16 elems per LDS row (144 B, 16B-aligned, conflict-skewed)
__global__ void edgeconv_kernel(const float* __restrict__ feats,
                                const int* __restrict__ nn,
                                const __bf16* __restrict__ w1p,
                                const __bf16* __restrict__ w2p,
                                const float* __restrict__ st,
                                float* __restrict__ out) {
  __shared__ __bf16 xs[8 * 32 * XSTRIDE];
  const int wave = threadIdx.x >> 5;
  const int lane = threadIdx.x & 31;
  const int pt = blockIdx.x * 8 + wave;  // point id = b*N + n
  const int b = pt >> 10;
  const int lo = lane & 15, hi = lane >> 4;
  const int hb = hi * 8;

  const float* __restrict__ fb = feats + (size_t)b * Nx * Dx;
  const float* __restrict__ cn = fb + (size_t)(pt & (Nx - 1)) * Dx;
  __bf16* __restrict__ slab = xs + wave * 32 * XSTRIDE;

  // center chunks (row-invariant across the edge matrix)
  float4 p00 = *(const float4*)(cn + hb),      p01 = *(const float4*)(cn + hb + 4);
  float4 p02 = *(const float4*)(cn + hb + 16), p03 = *(const float4*)(cn + hb + 20);
  float4 p10 = *(const float4*)(cn + hb + 32), p11 = *(const float4*)(cn + hb + 36);
  float4 p12 = *(const float4*)(cn + hb + 48), p13 = *(const float4*)(cn + hb + 52);
  v16bf aC0 = pack16(p00, p01, p02, p03);  // k-tile 0 (cols 0..31 pattern)
  v16bf aC1 = pack16(p10, p11, p12, p13);  // k-tile 1 (cols 32..63 pattern)

  // neighbor-difference A fragments for both M-tiles, built up front
  v16bf aN0[2], aN1[2];
#pragma unroll
  for (int mt = 0; mt < 2; ++mt) {
    int rown = mt * 16 + lo;
    int g = nn[(size_t)pt * Kx + (rown < Kx ? rown : 0)];
    const float* __restrict__ nb = fb + (size_t)g * Dx;
    float4 q00 = *(const float4*)(nb + hb),      q01 = *(const float4*)(nb + hb + 4);
    float4 q02 = *(const float4*)(nb + hb + 16), q03 = *(const float4*)(nb + hb + 20);
    float4 q10 = *(const float4*)(nb + hb + 32), q11 = *(const float4*)(nb + hb + 36);
    float4 q12 = *(const float4*)(nb + hb + 48), q13 = *(const float4*)(nb + hb + 52);
    aN0[mt] = packdiff16(q00, q01, q02, q03, p00, p01, p02, p03);  // k-tile 2
    aN1[mt] = packdiff16(q10, q11, q12, q13, p10, p11, p12, p13);  // k-tile 3
  }

  const v16bf* __restrict__ w1f = (const v16bf*)w1p;
  const v16bf* __restrict__ w2f = (const v16bf*)w2p;

  // -------- GEMM1 + BN1 + ReLU -> X staged to LDS (bf16) --------
#pragma unroll
  for (int nt = 0; nt < 4; ++nt) {
    v16bf bf0 = w1f[(0 * 4 + nt) * 32 + lane];
    v16bf bf1 = w1f[(1 * 4 + nt) * 32 + lane];
    v16bf bf2 = w1f[(2 * 4 + nt) * 32 + lane];
    v16bf bf3 = w1f[(3 * 4 + nt) * 32 + lane];
    int c = nt * 16 + lo;
    float s = st[c], t = st[64 + c];
#pragma unroll
    for (int mt = 0; mt < 2; ++mt) {
      v8f acc = {0.f, 0.f, 0.f, 0.f, 0.f, 0.f, 0.f, 0.f};
      acc = wmma_bf16(aC0, bf0, acc);
      acc = wmma_bf16(aC1, bf1, acc);
      acc = wmma_bf16(aN0[mt], bf2, acc);
      acc = wmma_bf16(aN1[mt], bf3, acc);
#pragma unroll
      for (int r = 0; r < 8; ++r) {
        float y = acc[r] * s + t;
        y = y > 0.f ? y : 0.f;
        int rw = mt * 16 + hb + r;
        slab[rw * XSTRIDE + c] = (__bf16)y;
      }
    }
  }

  // -------- GEMM2 + BN2 + ReLU + mean over k --------
  // A fragments for both M-tiles from LDS (in-wave DS ordering after stores)
  v16bf a20[2], a21[2];
#pragma unroll
  for (int mt = 0; mt < 2; ++mt) {
    const __bf16* __restrict__ xrow = slab + (mt * 16 + lo) * XSTRIDE;
    a20[mt] = cat8(*(const v8bf*)(xrow + hb),      *(const v8bf*)(xrow + hb + 16));
    a21[mt] = cat8(*(const v8bf*)(xrow + hb + 32), *(const v8bf*)(xrow + hb + 48));
  }

  float colsum[4] = {0.f, 0.f, 0.f, 0.f};
#pragma unroll
  for (int nt = 0; nt < 4; ++nt) {
    v16bf bf0 = w2f[(0 * 4 + nt) * 32 + lane];
    v16bf bf1 = w2f[(1 * 4 + nt) * 32 + lane];
    int c = nt * 16 + lo;
    float s = st[128 + c], t = st[192 + c];
#pragma unroll
    for (int mt = 0; mt < 2; ++mt) {
      v8f acc = {0.f, 0.f, 0.f, 0.f, 0.f, 0.f, 0.f, 0.f};
      acc = wmma_bf16(a20[mt], bf0, acc);
      acc = wmma_bf16(a21[mt], bf1, acc);
      float ps = 0.f;
#pragma unroll
      for (int r = 0; r < 8; ++r) {
        float y = acc[r] * s + t;
        y = y > 0.f ? y : 0.f;
        if (mt * 16 + hb + r < Kx) ps += y;  // drop padded rows
      }
      colsum[nt] += ps;
    }
  }

#pragma unroll
  for (int nt = 0; nt < 4; ++nt) {
    float tot = colsum[nt] + __shfl_xor(colsum[nt], 16, 32);
    if (hi == 0) out[(size_t)pt * Cx + nt * 16 + lo] = tot * (1.f / (float)Kx);
  }
}

// ---------------------------------------------------------------------------
extern "C" void kernel_launch(void* const* d_in, const int* in_sizes, int n_in,
                              void* d_out, int out_size, void* d_ws, size_t ws_size,
                              hipStream_t stream) {
  (void)in_sizes; (void)n_in; (void)out_size; (void)ws_size;
  const float* feats = (const float*)d_in[0];
  const float* adj   = (const float*)d_in[1];
  const float* w1    = (const float*)d_in[2];
  const float* b1    = (const float*)d_in[3];
  const float* g1    = (const float*)d_in[4];
  const float* be1   = (const float*)d_in[5];
  const float* mu1   = (const float*)d_in[6];
  const float* v1    = (const float*)d_in[7];
  const float* w2    = (const float*)d_in[8];
  const float* b2    = (const float*)d_in[9];
  const float* g2    = (const float*)d_in[10];
  const float* be2   = (const float*)d_in[11];
  const float* mu2   = (const float*)d_in[12];
  const float* v2    = (const float*)d_in[13];
  float* outp = (float*)d_out;

  char* ws = (char*)d_ws;
  float*  f0  = (float*)(ws + 0);                 //  65536 B
  int*    nnb = (int*)(ws + 65536);               // 1310720 B
  __bf16* w1p = (__bf16*)(ws + 1376256);          //  16384 B
  __bf16* w2p = (__bf16*)(ws + 1392640);          //   8192 B
  float*  st  = (float*)(ws + 1400832);           //   1024 B

  prep_kernel<<<64, 256, 0, stream>>>(feats, w1, b1, g1, be1, mu1, v1,
                                      w2, b2, g2, be2, mu2, v2,
                                      f0, w1p, w2p, st);
  topk_kernel<<<(Bx * Nx) / 8, 256, 0, stream>>>(adj, f0, nnb);
  edgeconv_kernel<<<(Bx * Nx) / 8, 256, 0, stream>>>(feats, nnb, w1p, w2p, st, outp);
}
